// ResidualExpansionGAT_87256555585813
// MI455X (gfx1250) — matrix operand order repaired and checked by reading, not compiled, not run on hardware
//
#include <hip/hip_runtime.h>
#include <stdint.h>

// ---------------- problem constants ----------------
#define N_NODES 50000
#define E_EDGES 500000
#define EE_TOT  (E_EDGES + N_NODES)   // with self-loops
#define H_HEADS 4
#define C_CH    128
#define D1      512
#define DIN     182
#define KPAD1   192                   // 182 padded to mult of 32
#define LN_EPSF 1e-5f
#define NEG_SLOPE 0.2f

typedef __bf16 bf16;
typedef __attribute__((ext_vector_type(16))) __bf16 v16bf;
typedef __attribute__((ext_vector_type(8)))  float  v8f;

static __device__ __forceinline__ void atomicMaxF(float* a, float v) {
    // float max via signed/unsigned int ordering trick (works for +/-, init -inf)
    if (v >= 0.0f) atomicMax((int*)a, __float_as_int(v));
    else           atomicMin((unsigned int*)a, __float_as_uint(v));
}

// ---------------- weight packing: W[K,Ncols] f32 -> B-fragment bf16 ----------------
// Layout: Wp[((kblk*Ncols + col)*32) + klocal], klocal = K offset within 32-chunk.
// Lane n (B fragment) then reads 8 bf16 at half*8 and 8 more at half*8+16.
__global__ void packW_kernel(const float* __restrict__ W, bf16* __restrict__ Wp,
                             int K, int Kpad, int Ncols, size_t total) {
    size_t i = (size_t)blockIdx.x * blockDim.x + threadIdx.x;
    if (i >= total) return;
    int kl = (int)(i & 31);
    size_t chunk = i >> 5;
    int col  = (int)(chunk % Ncols);
    int kblk = (int)(chunk / Ncols);
    int k = kblk * 32 + kl;
    float v = (k < K) ? W[(size_t)k * Ncols + col] : 0.0f;
    Wp[i] = (bf16)v;
}

// ---------------- activation staging: f32 [N,K] -> bf16 [N,Kpad] (zero pad) -------
__global__ void cvtA_kernel(const float* __restrict__ X, bf16* __restrict__ Y,
                            int K, int Kpad, size_t total) {
    size_t i = (size_t)blockIdx.x * blockDim.x + threadIdx.x;
    if (i >= total) return;
    int k = (int)(i % Kpad);
    size_t n = i / Kpad;
    Y[i] = (bf16)((k < K) ? X[n * K + k] : 0.0f);
}

// ---------------- bf16 WMMA GEMM: out = epilogue(A[N,Kpad] @ Wp + bias) -----------
// block = 128 (4 waves); each wave owns a 16-row x 64-col output strip
// (4 accumulators). A fragment loaded once per K-step, reused by 4 WMMAs.
// act: 0=none, 1=relu.  resid!=null: out = resid + scale*(AB+bias).
// exactly one of outF/outB is non-null.
__global__ __launch_bounds__(128)
void gemm_bf16_wmma(const bf16* __restrict__ A, const bf16* __restrict__ Wp,
                    const float* __restrict__ bias,
                    const float* __restrict__ resid, const float* __restrict__ scale_ptr,
                    float* __restrict__ outF, bf16* __restrict__ outB,
                    int Kpad, int Ncols, int act) {
    const int wave = threadIdx.x >> 5;
    const int lane = threadIdx.x & 31;
    const int row0 = blockIdx.x * 16;
    const int colS = (blockIdx.y * 4 + wave) * 64;   // 64-col strip
    const int m    = lane & 15;
    const int half = lane >> 4;

    const bf16* arow = A + (size_t)(row0 + m) * Kpad;

    v8f c0 = {}, c1 = {}, c2 = {}, c3 = {};
    const int nkb = Kpad >> 5;
#pragma unroll 2
    for (int kb = 0; kb < nkb; ++kb) {
        union { v16bf v; uint4 q[2]; } ua, ub0, ub1, ub2, ub3;
        const int k0 = kb * 32 + half * 8;
        ua.q[0] = *(const uint4*)(arow + k0);          // K = k0..k0+7
        ua.q[1] = *(const uint4*)(arow + k0 + 16);     // K = k0+16..k0+23

        const bf16* bbase = Wp + (((size_t)kb * Ncols + colS + m) << 5) + half * 8;
        // 4 column tiles, 16 cols apart => (16 << 5) bf16 apart in packed layout
        ub0.q[0] = *(const uint4*)(bbase);
        ub0.q[1] = *(const uint4*)(bbase + 16);
        ub1.q[0] = *(const uint4*)(bbase + (16 << 5));
        ub1.q[1] = *(const uint4*)(bbase + (16 << 5) + 16);
        ub2.q[0] = *(const uint4*)(bbase + (32 << 5));
        ub2.q[1] = *(const uint4*)(bbase + (32 << 5) + 16);
        ub3.q[0] = *(const uint4*)(bbase + (48 << 5));
        ub3.q[1] = *(const uint4*)(bbase + (48 << 5) + 16);

        c0 = __builtin_amdgcn_wmma_f32_16x16x32_bf16(false, ua.v, false, ub0.v, (short)0, c0, false, false);
        c1 = __builtin_amdgcn_wmma_f32_16x16x32_bf16(false, ua.v, false, ub1.v, (short)0, c1, false, false);
        c2 = __builtin_amdgcn_wmma_f32_16x16x32_bf16(false, ua.v, false, ub2.v, (short)0, c2, false, false);
        c3 = __builtin_amdgcn_wmma_f32_16x16x32_bf16(false, ua.v, false, ub3.v, (short)0, c3, false, false);
    }

    const float scale = scale_ptr ? scale_ptr[0] : 1.0f;
    v8f acc[4] = {c0, c1, c2, c3};
#pragma unroll
    for (int t = 0; t < 4; ++t) {
        const int colc = colS + t * 16 + m;
        const float bcol = bias ? bias[colc] : 0.0f;
#pragma unroll
        for (int r = 0; r < 8; ++r) {
            const int row = row0 + r + half * 8;
            float v = acc[t][r] + bcol;
            if (act == 1) v = v > 0.0f ? v : 0.0f;
            if (resid) v = resid[(size_t)row * Ncols + colc] + scale * v;
            const size_t oi = (size_t)row * Ncols + colc;
            if (outF) outF[oi] = v;
            if (outB) outB[oi] = (bf16)v;
        }
    }
}

// ---------------- per-(node,head) attention logits --------------------------------
__global__ void attn_logits_kernel(const float* __restrict__ Hm,
                                   const float* __restrict__ a_s, const float* __restrict__ a_d,
                                   float* __restrict__ als, float* __restrict__ ald) {
    int t = blockIdx.x * blockDim.x + threadIdx.x;
    if (t >= N_NODES * H_HEADS) return;
    const int h = t & 3, n = t >> 2;
    const float* hp = Hm + (size_t)n * D1 + h * C_CH;
    const float* ap = a_s + h * C_CH;
    const float* bp = a_d + h * C_CH;
    float s = 0.f, d = 0.f;
#pragma unroll 8
    for (int cI = 0; cI < C_CH; ++cI) { float v = hp[cI]; s += v * ap[cI]; d += v * bp[cI]; }
    als[t] = s; ald[t] = d;
}

// ---------------- aggregation init ------------------------------------------------
__global__ void init_agg_kernel(float* __restrict__ acc, float* __restrict__ mx,
                                float* __restrict__ den) {
    size_t i = (size_t)blockIdx.x * blockDim.x + threadIdx.x;
    if (i < (size_t)N_NODES * D1) acc[i] = 0.0f;
    if (i < (size_t)N_NODES * H_HEADS) { mx[i] = -__builtin_inff(); den[i] = 0.0f; }
}

// ---------------- edge pass 1: segment max ----------------------------------------
__global__ void edge_max_kernel(const int* __restrict__ ei,
                                const float* __restrict__ als, const float* __restrict__ ald,
                                float* __restrict__ mx) {
    size_t t = (size_t)blockIdx.x * blockDim.x + threadIdx.x;
    if (t >= (size_t)EE_TOT * H_HEADS) return;
    const int h = (int)(t & 3);
    const size_t e = t >> 2;
    int s_, d_;
    if (e < E_EDGES) { s_ = ei[e]; d_ = ei[E_EDGES + e]; }
    else             { s_ = d_ = (int)(e - E_EDGES); }
    float v = als[s_ * 4 + h] + ald[d_ * 4 + h];
    v = v > 0.0f ? v : NEG_SLOPE * v;
    atomicMaxF(&mx[d_ * 4 + h], v);
}

// ---------------- edge pass 2: exp / denom / weighted scatter ---------------------
// one wave per edge; each lane handles 16 channels (head = lane>>3)
__global__ __launch_bounds__(256)
void edge_acc_kernel(const int* __restrict__ ei,
                     const float* __restrict__ als, const float* __restrict__ ald,
                     const float* __restrict__ mx,
                     const float* __restrict__ Hm,
                     float* __restrict__ acc, float* __restrict__ den) {
    const int lane = threadIdx.x & 31;
    const size_t e = (size_t)blockIdx.x * 8 + (threadIdx.x >> 5);
    if (e >= EE_TOT) return;
    int s_, d_;
    if (e < E_EDGES) { s_ = ei[e]; d_ = ei[E_EDGES + e]; }
    else             { s_ = d_ = (int)(e - E_EDGES); }
    const int h = lane >> 3;
    float v = als[s_ * 4 + h] + ald[d_ * 4 + h];
    v = v > 0.0f ? v : NEG_SLOPE * v;
    const float ex = __expf(v - mx[d_ * 4 + h]);
    if ((lane & 7) == 0) atomicAdd(&den[d_ * 4 + h], ex);
    const float* hs = Hm + (size_t)s_ * D1 + lane * 16;
    float* ac = acc + (size_t)d_ * D1 + lane * 16;
#pragma unroll
    for (int i = 0; i < 16; i += 4) {
        const float4 hv = *(const float4*)(hs + i);
        atomicAdd(ac + i + 0, hv.x * ex);
        atomicAdd(ac + i + 1, hv.y * ex);
        atomicAdd(ac + i + 2, hv.z * ex);
        atomicAdd(ac + i + 3, hv.w * ex);
    }
}

// ---------------- finalize: /denom + bias, LayerNorm, ELU, -> bf16 ----------------
__global__ __launch_bounds__(512)
void finalize_ln_kernel(const float* __restrict__ acc, const float* __restrict__ den,
                        const float* __restrict__ bias,
                        const float* __restrict__ g, const float* __restrict__ bn,
                        bf16* __restrict__ out) {
    __shared__ float red[16];
    __shared__ float stat[2];
    const int n = blockIdx.x;
    const int c = threadIdx.x;
    const float dnm = den[n * 4 + (c >> 7)];
    const float v = acc[(size_t)n * D1 + c] / dnm + bias[c];

    float s = v;
#pragma unroll
    for (int o = 16; o > 0; o >>= 1) s += __shfl_down(s, o, 32);
    if ((c & 31) == 0) red[c >> 5] = s;
    __syncthreads();
    if (c == 0) { float t = 0.f; for (int i = 0; i < 16; ++i) t += red[i]; stat[0] = t * (1.0f / D1); }
    __syncthreads();
    const float mu = stat[0];
    const float dv = v - mu;
    float s2 = dv * dv;
    __syncthreads();               // red[] reuse
#pragma unroll
    for (int o = 16; o > 0; o >>= 1) s2 += __shfl_down(s2, o, 32);
    if ((c & 31) == 0) red[c >> 5] = s2;
    __syncthreads();
    if (c == 0) { float t = 0.f; for (int i = 0; i < 16; ++i) t += red[i]; stat[1] = rsqrtf(t * (1.0f / D1) + LN_EPSF); }
    __syncthreads();
    float y = dv * stat[1] * g[c] + bn[c];
    y = y > 0.0f ? y : (__expf(y) - 1.0f);   // ELU
    out[(size_t)n * D1 + c] = (bf16)y;
}

// ---------------- tiny classifier head: [N,512] bf16 @ [512,2] + bc2 --------------
__global__ void classifier_kernel(const bf16* __restrict__ Hb,
                                  const float* __restrict__ Wc2, const float* __restrict__ bc2,
                                  float* __restrict__ out) {
    int n = blockIdx.x * blockDim.x + threadIdx.x;
    if (n >= N_NODES) return;
    const bf16* hp = Hb + (size_t)n * D1;
    float a0 = bc2[0], a1 = bc2[1];
#pragma unroll 8
    for (int k = 0; k < D1; ++k) {
        const float hv = (float)hp[k];
        a0 += hv * Wc2[k * 2 + 0];
        a1 += hv * Wc2[k * 2 + 1];
    }
    out[n * 2 + 0] = a0;
    out[n * 2 + 1] = a1;
}

// =================================================================================
static inline size_t cdiv(size_t a, size_t b) { return (a + b - 1) / b; }

extern "C" void kernel_launch(void* const* d_in, const int* in_sizes, int n_in,
                              void* d_out, int out_size, void* d_ws, size_t ws_size,
                              hipStream_t stream) {
    (void)in_sizes; (void)n_in; (void)out_size; (void)ws_size;
    const float* x    = (const float*)d_in[0];
    const int*   ei   = (const int*)  d_in[1];
    const float* W1   = (const float*)d_in[2];
    const float* as1  = (const float*)d_in[3];
    const float* ad1  = (const float*)d_in[4];
    const float* b1   = (const float*)d_in[5];
    const float* g1   = (const float*)d_in[6];
    const float* bn1  = (const float*)d_in[7];
    const float* We1  = (const float*)d_in[8];
    const float* bE1  = (const float*)d_in[9];
    const float* We2  = (const float*)d_in[10];
    const float* bE2  = (const float*)d_in[11];
    const float* Wr   = (const float*)d_in[12];
    const float* br   = (const float*)d_in[13];
    const float* rsc  = (const float*)d_in[14];
    const float* W2   = (const float*)d_in[15];
    const float* as2  = (const float*)d_in[16];
    const float* ad2  = (const float*)d_in[17];
    const float* b2   = (const float*)d_in[18];
    const float* g2   = (const float*)d_in[19];
    const float* bn2  = (const float*)d_in[20];
    const float* Wc1  = (const float*)d_in[21];
    const float* bc1  = (const float*)d_in[22];
    const float* Wc2  = (const float*)d_in[23];
    const float* bc2  = (const float*)d_in[24];

    // ---- workspace carve-up ----
    char* base = (char*)d_ws;
    size_t off = 0;
    auto carve = [&](size_t bytes) { void* p = base + off; off = (off + bytes + 255) & ~(size_t)255; return p; };
    float* H1   = (float*)carve((size_t)N_NODES * D1 * 4);      // pre-agg h / post-GAT scratch
    float* ACC  = (float*)carve((size_t)N_NODES * D1 * 4);      // aggregation acc / MLP residual R
    bf16*  ABF  = (bf16*) carve((size_t)N_NODES * D1 * 2);      // bf16 activation staging A
    bf16*  BBF  = (bf16*) carve((size_t)N_NODES * D1 * 2);      // bf16 activation staging B
    float* ALS  = (float*)carve((size_t)N_NODES * H_HEADS * 4);
    float* ALD  = (float*)carve((size_t)N_NODES * H_HEADS * 4);
    float* MX   = (float*)carve((size_t)N_NODES * H_HEADS * 4);
    float* DEN  = (float*)carve((size_t)N_NODES * H_HEADS * 4);
    bf16*  W1P  = (bf16*) carve((size_t)(KPAD1 / 32) * D1  * 32 * 2);
    bf16*  WE1P = (bf16*) carve((size_t)(D1 / 32)    * D1  * 32 * 2);
    bf16*  WE2P = (bf16*) carve((size_t)(D1 / 32)    * 256 * 32 * 2);
    bf16*  WRP  = (bf16*) carve((size_t)(D1 / 32)    * 256 * 32 * 2);
    bf16*  W2P  = (bf16*) carve((size_t)(256 / 32)   * D1  * 32 * 2);
    bf16*  WC1P = (bf16*) carve((size_t)(D1 / 32)    * D1  * 32 * 2);

    const int TB = 256;
    auto packW = [&](const float* W, bf16* Wp, int K, int Kpad, int Nc) {
        size_t tot = (size_t)(Kpad / 32) * Nc * 32;
        packW_kernel<<<cdiv(tot, TB), TB, 0, stream>>>(W, Wp, K, Kpad, Nc, tot);
    };
    packW(W1,  W1P,  DIN, KPAD1, D1);
    packW(We1, WE1P, D1,  D1,    D1);
    packW(We2, WE2P, D1,  D1,    256);
    packW(Wr,  WRP,  D1,  D1,    256);
    packW(W2,  W2P,  256, 256,   D1);
    packW(Wc1, WC1P, D1,  D1,    D1);

    const dim3 gemmBlk(128);
    const dim3 grid512(N_NODES / 16, D1 / 256);   // 3125 x 2  (4 waves x 64 cols = 256/block)
    const dim3 grid256(N_NODES / 16, 256 / 256);  // 3125 x 1

    auto run_gat = [&](const float* a_s, const float* a_d, const float* bias,
                       const float* g, const float* bn) {
        attn_logits_kernel<<<cdiv((size_t)N_NODES * H_HEADS, TB), TB, 0, stream>>>(H1, a_s, a_d, ALS, ALD);
        init_agg_kernel<<<cdiv((size_t)N_NODES * D1, TB), TB, 0, stream>>>(ACC, MX, DEN);
        edge_max_kernel<<<cdiv((size_t)EE_TOT * H_HEADS, TB), TB, 0, stream>>>(ei, ALS, ALD, MX);
        edge_acc_kernel<<<cdiv((size_t)EE_TOT, 8), 256, 0, stream>>>(ei, ALS, ALD, MX, H1, ACC, DEN);
        finalize_ln_kernel<<<N_NODES, D1, 0, stream>>>(ACC, DEN, bias, g, bn, ABF);
    };

    // ---- Stage A: GAT layer 1 (x:182 -> 4x128) ----
    {
        size_t tot = (size_t)N_NODES * KPAD1;
        cvtA_kernel<<<cdiv(tot, TB), TB, 0, stream>>>(x, ABF, DIN, KPAD1, tot);
    }
    gemm_bf16_wmma<<<grid512, gemmBlk, 0, stream>>>(ABF, W1P, nullptr, nullptr, nullptr,
                                                    H1, nullptr, KPAD1, D1, 0);
    run_gat(as1, ad1, b1, g1, bn1);          // -> ABF = elu(ln(h1)) bf16 [N,512]

    // ---- Stage B: expansion MLP + residual combine ----
    // R = ABF @ Wr + br            -> ACC (f32 [N,256])
    gemm_bf16_wmma<<<grid256, gemmBlk, 0, stream>>>(ABF, WRP, br, nullptr, nullptr,
                                                    ACC, nullptr, D1, 256, 0);
    // T = relu(ABF @ We1 + bE1)    -> BBF (bf16 [N,512])
    gemm_bf16_wmma<<<grid512, gemmBlk, 0, stream>>>(ABF, WE1P, bE1, nullptr, nullptr,
                                                    nullptr, BBF, D1, D1, 1);
    // h_comb = R + rscale*(T @ We2 + bE2) -> ABF (bf16 [N,256])
    gemm_bf16_wmma<<<grid256, gemmBlk, 0, stream>>>(BBF, WE2P, bE2, ACC, rsc,
                                                    nullptr, ABF, D1, 256, 0);

    // ---- Stage C: GAT layer 2 (256 -> 4x128) ----
    gemm_bf16_wmma<<<grid512, gemmBlk, 0, stream>>>(ABF, W2P, nullptr, nullptr, nullptr,
                                                    H1, nullptr, 256, D1, 0);
    run_gat(as2, ad2, b2, g2, bn2);          // -> ABF = elu(ln(h2)) bf16 [N,512]

    // ---- Stage D: classifier ----
    gemm_bf16_wmma<<<grid512, gemmBlk, 0, stream>>>(ABF, WC1P, bc1, nullptr, nullptr,
                                                    nullptr, BBF, D1, D1, 1);
    classifier_kernel<<<cdiv(N_NODES, TB), TB, 0, stream>>>(BBF, Wc2, bc2, (float*)d_out);
}
